// DSnetwork_627065225441
// MI455X (gfx1250) — matrix operand order repaired and checked by reading, not compiled, not run on hardware
//
#include <hip/hip_runtime.h>

#define N_SUBG     100000
#define NUM_GRAPHS 4096
#define DDIM       512
#define NUM_TASKS  10

typedef __bf16 bf16_t;
typedef __attribute__((ext_vector_type(16))) __bf16 v16bf;
typedef __attribute__((ext_vector_type(8)))  __bf16 v8bf;
typedef __attribute__((ext_vector_type(8)))  float  v8f;
typedef __attribute__((ext_vector_type(4)))  float  v4f;

#define TILE_M 128
#define TILE_N 128
#define TILE_K 32
#define LDSS   40   // bf16 elems per LDS row: 32 + 8 pad -> 80 bytes (16B aligned rows)

// ---------------------------------------------------------------------------
// Tiled WMMA GEMM, register-staged double buffering (1 barrier / K-step):
//   C[M,N] = A[M,K] * BT[N,K]^T  (+ epilogue by MODE)
//   MODE 0: C = A@B + bias                      -> f32 out   (graph-term GEMM)
//   MODE 1: C = elu(A@B + bias + x2[idx[m]])    -> bf16 out  (per-subgraph layer)
//   MODE 2: C = relu(A@B + bias)                -> f32 out   (readout fc1)
// A dtype: float (converted to bf16 on the fly) or bf16.
// ---------------------------------------------------------------------------
template<typename TA, int MODE>
__global__ __launch_bounds__(256) void gemm_wmma(
    const TA* __restrict__ A, const bf16_t* __restrict__ BT,
    const float* __restrict__ bias, const float* __restrict__ x2,
    const int* __restrict__ idx, void* __restrict__ Cout,
    int M, int N, int K)
{
  __shared__ __align__(16) bf16_t lA[2][TILE_M * LDSS];
  __shared__ __align__(16) bf16_t lB[2][TILE_N * LDSS];

  const int tid  = threadIdx.x;
  const int lane = tid & 31;
  const int wv   = tid >> 5;      // 8 waves
  const int wm   = wv & 3;        // 4 wave rows  (32 rows each)
  const int wn   = wv >> 2;       // 2 wave cols  (64 cols each)
  const int half = lane >> 4;     // lane half (WMMA layout)
  const int l16  = lane & 15;
  const int frow = tid >> 1;      // LDS fill: row 0..127
  const int fseg = tid & 1;       // LDS fill: 16-elem K segment

  const long m0 = (long)blockIdx.y * TILE_M;
  const long n0 = (long)blockIdx.x * TILE_N;
  const long gmA = m0 + frow;     // A source row for this thread's fill
  const bool aInR = gmA < M;

  const v8f vzero = {0.f,0.f,0.f,0.f,0.f,0.f,0.f,0.f};
  v8f acc[2][4];
#pragma unroll
  for (int t = 0; t < 2; t++)
#pragma unroll
    for (int j = 0; j < 4; j++) acc[t][j] = vzero;

  // staging registers (global -> regs -> LDS)
  v8bf stA0, stA1, stB0, stB1;

  auto load_stage = [&](int kt) {
    if (sizeof(TA) == 4) {
      if (aInR) {
        const v4f* s = (const v4f*)((const float*)A + gmA * K + kt + fseg * 16);
        v4f x0 = s[0], x1 = s[1], x2v = s[2], x3 = s[3];
#pragma unroll
        for (int j = 0; j < 4; j++) {
          stA0[j]   = (bf16_t)x0[j]; stA0[4+j] = (bf16_t)x1[j];
          stA1[j]   = (bf16_t)x2v[j]; stA1[4+j] = (bf16_t)x3[j];
        }
      } else {
        const bf16_t z = (bf16_t)0.f;
#pragma unroll
        for (int j = 0; j < 8; j++) { stA0[j] = z; stA1[j] = z; }
      }
    } else {
      if (aInR) {
        const v8bf* s = (const v8bf*)((const bf16_t*)A + gmA * K + kt + fseg * 16);
        stA0 = s[0]; stA1 = s[1];
      } else {
        const bf16_t z = (bf16_t)0.f;
#pragma unroll
        for (int j = 0; j < 8; j++) { stA0[j] = z; stA1[j] = z; }
      }
    }
    const v8bf* sb = (const v8bf*)(BT + (n0 + frow) * K + kt + fseg * 16);
    stB0 = sb[0]; stB1 = sb[1];
  };

  auto store_stage = [&](int b) {
    bf16_t* da = &lA[b][frow * LDSS + fseg * 16];
    ((v8bf*)da)[0] = stA0; ((v8bf*)da)[1] = stA1;
    bf16_t* db = &lB[b][frow * LDSS + fseg * 16];
    ((v8bf*)db)[0] = stB0; ((v8bf*)db)[1] = stB1;
  };

  // prologue: tile 0 into buffer 0
  load_stage(0);
  store_stage(0);
  __syncthreads();

  int buf = 0;
  for (int kt = 0; kt < K; kt += TILE_K) {
    // prefetch next tile into the other buffer while computing on this one
    if (kt + TILE_K < K) {
      load_stage(kt + TILE_K);   // global_load (overlaps WMMA below)
      store_stage(buf ^ 1);      // ds_store gated only on the global loads
    }

    // ---- per-wave fragment loads (ISA 16-bit A/B VGPR layouts) ----
    v16bf afr[2], bfr[4];
#pragma unroll
    for (int t = 0; t < 2; t++) {
      const bf16_t* p = &lA[buf][(wm * 32 + t * 16 + l16) * LDSS + half * 8];
      union { v16bf v; v8bf h[2]; } u;
      u.h[0] = *(const v8bf*)p;          // K = 8*half .. +7
      u.h[1] = *(const v8bf*)(p + 16);   // K = 8*half+16 .. +7
      afr[t] = u.v;
    }
#pragma unroll
    for (int j = 0; j < 4; j++) {
      const bf16_t* p = &lB[buf][(wn * 64 + j * 16 + l16) * LDSS + half * 8];
      union { v16bf v; v8bf h[2]; } u;
      u.h[0] = *(const v8bf*)p;          // K = 8*half .. +7
      u.h[1] = *(const v8bf*)(p + 16);   // K = 8*half+16 .. +7
      bfr[j] = u.v;
    }
#pragma unroll
    for (int t = 0; t < 2; t++)
#pragma unroll
      for (int j = 0; j < 4; j++)
        acc[t][j] = __builtin_amdgcn_wmma_f32_16x16x32_bf16(
            false, afr[t], false, bfr[j], (short)0, acc[t][j], false, false);

    // one barrier: readers of lX[buf] done AND writers of lX[buf^1] done
    __syncthreads();
    buf ^= 1;
  }

  // ---- epilogue ----
#pragma unroll
  for (int t = 0; t < 2; t++) {
#pragma unroll
    for (int j = 0; j < 4; j++) {
      const long n = n0 + wn * 64 + j * 16 + l16;
      const float bn = bias[n];
#pragma unroll
      for (int r = 0; r < 8; r++) {
        const long m = m0 + wm * 32 + t * 16 + half * 8 + r;
        if (m < M) {
          float v = acc[t][j][r] + bn;
          if (MODE == 1) {
            v += x2[(long)idx[m] * N + n];
            v = v > 0.f ? v : (__expf(v) - 1.f);            // ELU
            ((bf16_t*)Cout)[m * N + n] = (bf16_t)v;
          } else if (MODE == 2) {
            ((float*)Cout)[m * N + n] = v > 0.f ? v : 0.f;  // ReLU
          } else {
            ((float*)Cout)[m * N + n] = v;
          }
        }
      }
    }
  }
}

// ---------------------------------------------------------------------------
// Segment sum (idx sorted): each thread run-length accumulates 16 rows of one
// column, one atomicAdd per segment boundary.
// ---------------------------------------------------------------------------
template<typename TA>
__global__ void segsum_kernel(const TA* __restrict__ h, const int* __restrict__ idx,
                              float* __restrict__ g, int M)
{
  const int c  = blockIdx.x * blockDim.x + threadIdx.x;  // 0..DDIM-1
  const int r0 = blockIdx.y * 16;
  int rend = r0 + 16; if (rend > M) rend = M;
  float a = 0.f;
  int cur = idx[r0];
  for (int r = r0; r < rend; r++) {
    int gi = idx[r];
    if (gi != cur) { atomicAdd(&g[(long)cur * DDIM + c], a); a = 0.f; cur = gi; }
    a += (float)h[(long)r * DDIM + c];
  }
  atomicAdd(&g[(long)cur * DDIM + c], a);
}

__global__ void count_kernel(const int* __restrict__ idx, float* __restrict__ cnt, int M) {
  int i = blockIdx.x * blockDim.x + threadIdx.x;
  if (i < M) atomicAdd(&cnt[idx[i]], 1.f);
}

__global__ void mean_kernel(float* __restrict__ g, const float* __restrict__ cnt) {
  int i = blockIdx.x * blockDim.x + threadIdx.x;   // NUM_GRAPHS*DDIM threads
  float c = cnt[i / DDIM];
  g[i] = g[i] / fmaxf(c, 1.f);
}

__global__ void zero_kernel(float* __restrict__ p, int n) {
  int i = blockIdx.x * blockDim.x + threadIdx.x;
  if (i < n) p[i] = 0.f;
}

// W[K][N] f32 -> WT[N][K] bf16
__global__ void convw_kernel(const float* __restrict__ W, bf16_t* __restrict__ WT,
                             int K, int N) {
  int i = blockIdx.x * blockDim.x + threadIdx.x;
  if (i >= K * N) return;
  int k = i / N, n = i - k * N;
  WT[(long)n * K + k] = (bf16_t)W[i];
}

// out[4096,10] = t[4096,1024] @ Wf2[1024,10] + bf2
__global__ void head_kernel(const float* __restrict__ tbuf, const float* __restrict__ W2,
                            const float* __restrict__ b2, float* __restrict__ out) {
  int o = blockIdx.x * blockDim.x + threadIdx.x;
  if (o >= NUM_GRAPHS * NUM_TASKS) return;
  int gidx = o / NUM_TASKS, task = o - gidx * NUM_TASKS;
  float s = b2[task];
  const float* trow = tbuf + (long)gidx * (2 * DDIM);
  for (int k = 0; k < 2 * DDIM; k++) s += trow[k] * W2[k * NUM_TASKS + task];
  out[o] = s;
}

// ---------------------------------------------------------------------------
extern "C" void kernel_launch(void* const* d_in, const int* in_sizes, int n_in,
                              void* d_out, int out_size, void* d_ws, size_t ws_size,
                              hipStream_t stream)
{
  (void)in_sizes; (void)n_in; (void)out_size; (void)ws_size;
  const float* h0   = (const float*)d_in[0];
  const int*   idx  = (const int*)  d_in[1];
  const float* Wfc  = (const float*)d_in[2];
  const float* bfc  = (const float*)d_in[3];
  const float* Wsum = (const float*)d_in[4];
  const float* bsum = (const float*)d_in[5];
  const float* Wf1  = (const float*)d_in[6];
  const float* bf1  = (const float*)d_in[7];
  const float* Wf2  = (const float*)d_in[8];
  const float* bf2  = (const float*)d_in[9];
  float* out = (float*)d_out;

  char* ws = (char*)d_ws;
  auto take = [&](size_t bytes) { char* p = ws; ws += (bytes + 255) & ~(size_t)255; return p; };

  bf16_t* WfcT  = (bf16_t*)take((size_t)3 * DDIM * DDIM * sizeof(bf16_t));
  bf16_t* WsumT = (bf16_t*)take((size_t)3 * DDIM * DDIM * sizeof(bf16_t));
  bf16_t* Wf1T  = (bf16_t*)take((size_t)2 * DDIM * DDIM * sizeof(bf16_t));
  float*  g     = (float*) take((size_t)NUM_GRAPHS * DDIM * sizeof(float));
  float*  cnt   = (float*) take((size_t)NUM_GRAPHS * sizeof(float));
  float*  x2    = (float*) take((size_t)NUM_GRAPHS * DDIM * sizeof(float));
  float*  tbuf  = (float*) take((size_t)NUM_GRAPHS * 2 * DDIM * sizeof(float));
  bf16_t* hb1   = (bf16_t*)take((size_t)N_SUBG * DDIM * sizeof(bf16_t));
  bf16_t* hb2   = (bf16_t*)take((size_t)N_SUBG * DDIM * sizeof(bf16_t));

  // --- weight prep: f32 -> transposed bf16 (cheap, every call: deterministic) ---
  {
    const int nel = DDIM * DDIM;
    dim3 b(256), gr((nel + 255) / 256);
    for (int l = 0; l < 3; l++) {
      convw_kernel<<<gr, b, 0, stream>>>(Wfc  + (size_t)l * nel, WfcT  + (size_t)l * nel, DDIM, DDIM);
      convw_kernel<<<gr, b, 0, stream>>>(Wsum + (size_t)l * nel, WsumT + (size_t)l * nel, DDIM, DDIM);
    }
    const int nel2 = DDIM * 2 * DDIM;
    convw_kernel<<<dim3((nel2 + 255) / 256), b, 0, stream>>>(Wf1, Wf1T, DDIM, 2 * DDIM);
  }

  // --- segment counts (idx is constant across layers) ---
  zero_kernel <<<dim3((NUM_GRAPHS + 255) / 256), dim3(256), 0, stream>>>(cnt, NUM_GRAPHS);
  count_kernel<<<dim3((N_SUBG + 255) / 256),     dim3(256), 0, stream>>>(idx, cnt, N_SUBG);

  const dim3 segGrid(DDIM / 128, (N_SUBG + 15) / 16), segBlk(128);
  const dim3 meanGrid((NUM_GRAPHS * DDIM) / 256), meanBlk(256);
  const dim3 gemmBlk(256);
  const dim3 gGridSmall(DDIM / TILE_N, NUM_GRAPHS / TILE_M);          // 4 x 32
  const dim3 gGridBig(DDIM / TILE_N, (N_SUBG + TILE_M - 1) / TILE_M); // 4 x 782
  const dim3 gGridF1((2 * DDIM) / TILE_N, NUM_GRAPHS / TILE_M);       // 8 x 32

  bf16_t* cur_in = nullptr;
  for (int l = 0; l < 3; l++) {
    zero_kernel<<<dim3((NUM_GRAPHS * DDIM + 255) / 256), dim3(256), 0, stream>>>(g, NUM_GRAPHS * DDIM);
    if (l == 0)
      segsum_kernel<float ><<<segGrid, segBlk, 0, stream>>>(h0,     idx, g, N_SUBG);
    else
      segsum_kernel<bf16_t><<<segGrid, segBlk, 0, stream>>>(cur_in, idx, g, N_SUBG);
    mean_kernel<<<meanGrid, meanBlk, 0, stream>>>(g, cnt);

    // x2 = mean(h) @ Wsum[l] + bsum[l]
    gemm_wmma<float, 0><<<gGridSmall, gemmBlk, 0, stream>>>(
        g, WsumT + (size_t)l * DDIM * DDIM, bsum + (size_t)l * DDIM,
        nullptr, nullptr, x2, NUM_GRAPHS, DDIM, DDIM);

    // h = elu(h @ Wfc[l] + bfc[l] + x2[idx])  -> bf16
    bf16_t* hout = (l == 1) ? hb2 : hb1;
    if (l == 0)
      gemm_wmma<float, 1><<<gGridBig, gemmBlk, 0, stream>>>(
          h0, WfcT, bfc, x2, idx, hout, N_SUBG, DDIM, DDIM);
    else
      gemm_wmma<bf16_t, 1><<<gGridBig, gemmBlk, 0, stream>>>(
          cur_in, WfcT + (size_t)l * DDIM * DDIM, bfc + (size_t)l * DDIM,
          x2, idx, hout, N_SUBG, DDIM, DDIM);
    cur_in = hout;
  }

  // --- readout ---
  zero_kernel<<<dim3((NUM_GRAPHS * DDIM + 255) / 256), dim3(256), 0, stream>>>(g, NUM_GRAPHS * DDIM);
  segsum_kernel<bf16_t><<<segGrid, segBlk, 0, stream>>>(cur_in, idx, g, N_SUBG);
  mean_kernel<<<meanGrid, meanBlk, 0, stream>>>(g, cnt);
  gemm_wmma<float, 2><<<gGridF1, gemmBlk, 0, stream>>>(
      g, Wf1T, bf1, nullptr, nullptr, tbuf, NUM_GRAPHS, 2 * DDIM, DDIM);
  head_kernel<<<dim3((NUM_GRAPHS * NUM_TASKS + 255) / 256), dim3(256), 0, stream>>>(
      tbuf, Wf2, bf2, out);
}
